// DynamicAttentionFusionLayer_48172353192360
// MI455X (gfx1250) — matrix operand order repaired and checked by reading, not compiled, not run on hardware
//
#include <hip/hip_runtime.h>
#include <hip/hip_bf16.h>

// ---------------------------------------------------------------------------
// DynamicAttentionFusionLayer for MI455X (gfx1250, wave32, WMMA).
// Dense GEMMs on v_wmma_f32_16x16x32_bf16 (f32 accumulate), 4 N-tiles/wave.
// ---------------------------------------------------------------------------

typedef __attribute__((ext_vector_type(16))) __bf16 v16bf;
typedef __attribute__((ext_vector_type(8)))  __bf16 v8bf;
typedef __attribute__((ext_vector_type(8)))  float  v8f;

#define HID_   128
#define CB_    768
#define HEADS_ 4

__device__ __forceinline__ unsigned short f2bf(float f) {
    unsigned u = __float_as_uint(f);
    unsigned r = u + 0x7FFFu + ((u >> 16) & 1u);   // round-to-nearest-even
    return (unsigned short)(r >> 16);
}
__device__ __forceinline__ unsigned f2ord(float f) {  // monotone float->uint
    unsigned u = __float_as_uint(f);
    return (u & 0x80000000u) ? ~u : (u | 0x80000000u);
}
__device__ __forceinline__ float ord2f(unsigned s) {
    return __uint_as_float((s & 0x80000000u) ? (s ^ 0x80000000u) : ~s);
}

// ------------------------- elementwise f32 -> bf16 -------------------------
__global__ void k_cvt_bf16(const float* __restrict__ src,
                           unsigned short* __restrict__ dst, long n) {
    long i = (long)blockIdx.x * 256 + threadIdx.x;
    if (i < n) dst[i] = f2bf(src[i]);
}

// ---- pack weight [K x (N1+N2)] (optionally two concatenated sources) into
// ---- the WMMA B-operand layout: ((kb/32)*nTiles + nb)*32*16 + lane*16 + e,
// ---- where K = kb + 8*(lane>=16) + (e&7) + 16*(e>>3), col = nb*16 + (lane&15).
__global__ void k_pack_w(const float* __restrict__ W1, const float* __restrict__ W2,
                         int K, int N1, int N2, unsigned short* __restrict__ out) {
    int N = N1 + N2;
    long total = (long)K * N;
    long idx = (long)blockIdx.x * 256 + threadIdx.x;
    if (idx >= total) return;
    int e    = (int)(idx & 15);
    int lane = (int)((idx >> 4) & 31);
    long rest = idx >> 9;
    int nTiles = N >> 4;
    int nb   = (int)(rest % nTiles);
    int kb32 = (int)(rest / nTiles);
    int k   = kb32 * 32 + ((lane >> 4) << 3) + (e & 7) + ((e >> 3) << 4);
    int col = nb * 16 + (lane & 15);
    float v = (col < N1) ? W1[(size_t)k * N1 + col]
                         : W2[(size_t)k * N2 + (col - N1)];
    out[idx] = f2bf(v);
}

__global__ void k_cat_bias(const float* __restrict__ b1, const float* __restrict__ b2,
                           float* __restrict__ out) {
    int t = threadIdx.x;                 // 256 threads
    out[t] = (t < HID_) ? b1[t] : b2[t - HID_];
}

// ----------------------- generic bf16 WMMA GEMM ----------------------------
// C[M x Nout] = A[M x K](bf16, row-major, stride lda) @ Wp(packed) + bias
// one wave computes a 16x64 tile (4 N-tiles) so each A fragment feeds 4 WMMAs
__global__ void k_gemm_bf16(const unsigned short* __restrict__ A, int lda,
                            const unsigned short* __restrict__ Wp,
                            const float* __restrict__ bias,
                            float* __restrict__ C, int ldc, int K, int Nout) {
    const int tileM  = blockIdx.x;
    const int tileN0 = blockIdx.y * 4;       // 4 consecutive N-tiles
    const int lane   = threadIdx.x;
    const int hs = lane >> 4, lid = lane & 15;
    const unsigned short* Arow = A + (size_t)(tileM * 16 + lid) * lda;
    const long bStep = (long)(Nout >> 4) * 512;
    const unsigned short* Bp0 = Wp + ((size_t)(tileN0 + 0) * 32 + lane) * 16;
    const unsigned short* Bp1 = Wp + ((size_t)(tileN0 + 1) * 32 + lane) * 16;
    const unsigned short* Bp2 = Wp + ((size_t)(tileN0 + 2) * 32 + lane) * 16;
    const unsigned short* Bp3 = Wp + ((size_t)(tileN0 + 3) * 32 + lane) * 16;
    v8f acc0 = {}, acc1 = {}, acc2 = {}, acc3 = {};
    for (int kb = 0; kb < K; kb += 32) {
        v8bf lo = *(const v8bf*)(Arow + kb + hs * 8);
        v8bf hi = *(const v8bf*)(Arow + kb + hs * 8 + 16);
        v16bf a = __builtin_shufflevector(lo, hi, 0,1,2,3,4,5,6,7,8,9,10,11,12,13,14,15);
        v16bf b0 = *(const v16bf*)Bp0;  Bp0 += bStep;
        v16bf b1 = *(const v16bf*)Bp1;  Bp1 += bStep;
        v16bf b2 = *(const v16bf*)Bp2;  Bp2 += bStep;
        v16bf b3 = *(const v16bf*)Bp3;  Bp3 += bStep;
        acc0 = __builtin_amdgcn_wmma_f32_16x16x32_bf16(false, a, false, b0, (short)0, acc0, false, false);
        acc1 = __builtin_amdgcn_wmma_f32_16x16x32_bf16(false, a, false, b1, (short)0, acc1, false, false);
        acc2 = __builtin_amdgcn_wmma_f32_16x16x32_bf16(false, a, false, b2, (short)0, acc2, false, false);
        acc3 = __builtin_amdgcn_wmma_f32_16x16x32_bf16(false, a, false, b3, (short)0, acc3, false, false);
    }
    const int rowBase = tileM * 16 + hs * 8;
#pragma unroll
    for (int j = 0; j < 4; ++j) {
        const v8f& acc = (j == 0) ? acc0 : (j == 1) ? acc1 : (j == 2) ? acc2 : acc3;
        int col = (tileN0 + j) * 16 + lid;
        float bv = bias ? bias[col] : 0.f;
#pragma unroll
        for (int r = 0; r < 8; ++r)
            C[(size_t)(rowBase + r) * ldc + col] = acc[r] + bv;
    }
}

// -------- GAT attention coefficients: a_src/a_dst = sum_c xp*att ----------
__global__ void k_attn_coef(const float* __restrict__ xp,
                            const float* __restrict__ att_src,
                            const float* __restrict__ att_dst,
                            float* __restrict__ as_, float* __restrict__ ad_) {
    __shared__ float ss[128], sd[128];
    int n = blockIdx.x, t = threadIdx.x;
    int h = t >> 5, l = t & 31;
    const float* xph = xp + (size_t)n * 512 + h * HID_;
    float ps = 0.f, pd = 0.f;
#pragma unroll
    for (int j = 0; j < 4; ++j) {
        float xv = xph[l + 32 * j];
        ps += xv * att_src[h * HID_ + l + 32 * j];
        pd += xv * att_dst[h * HID_ + l + 32 * j];
    }
    ss[t] = ps; sd[t] = pd;
    __syncthreads();
    for (int s = 16; s > 0; s >>= 1) {
        if (l < s) { ss[t] += ss[t + s]; sd[t] += sd[t + s]; }
        __syncthreads();
    }
    if (l == 0) { as_[n * 4 + h] = ss[t]; ad_[n * 4 + h] = sd[t]; }
}

// ---------------- edge softmax pass 1: logits + segment max ----------------
__global__ void k_edge_logits(const int* __restrict__ esrc, const int* __restrict__ edst,
                              int E, int Etot,
                              const float* __restrict__ as_, const float* __restrict__ ad_,
                              float* __restrict__ eval, unsigned* __restrict__ emax) {
    int t = blockIdx.x * 256 + threadIdx.x;
    if (t >= Etot) return;
    int s = (t < E) ? esrc[t] : (t - E);
    int d = (t < E) ? edst[t] : (t - E);
#pragma unroll
    for (int h = 0; h < HEADS_; ++h) {
        float e = as_[s * 4 + h] + ad_[d * 4 + h];
        e = (e > 0.f) ? e : 0.2f * e;                 // leaky_relu
        eval[(size_t)t * 4 + h] = e;
        atomicMax(&emax[d * 4 + h], f2ord(e));
    }
}

// ---------------- edge softmax pass 2: exp + segment sum -------------------
__global__ void k_edge_exp(const int* __restrict__ esrc, const int* __restrict__ edst,
                           int E, int Etot, const unsigned* __restrict__ emax,
                           float* __restrict__ eval, float* __restrict__ denom) {
    int t = blockIdx.x * 256 + threadIdx.x;
    if (t >= Etot) return;
    int d = (t < E) ? edst[t] : (t - E);
#pragma unroll
    for (int h = 0; h < HEADS_; ++h) {
        float ex = __expf(eval[(size_t)t * 4 + h] - ord2f(emax[d * 4 + h]));
        eval[(size_t)t * 4 + h] = ex;
        atomicAdd(&denom[d * 4 + h], ex);
    }
}

// --------- pass 3: alpha-weighted message scatter (f32 L2 atomics) ---------
__global__ void k_edge_scatter(const int* __restrict__ esrc, const int* __restrict__ edst,
                               int E, const float* __restrict__ eval,
                               const float* __restrict__ denom,
                               const float* __restrict__ xp, float* __restrict__ acc) {
    int t = blockIdx.x;           // edge id (grid = Etot)
    int c = threadIdx.x;          // 0..127
    int s = (t < E) ? esrc[t] : (t - E);
    int d = (t < E) ? edst[t] : (t - E);
#pragma unroll
    for (int h = 0; h < HEADS_; ++h) {
        float alpha = eval[(size_t)t * 4 + h] / denom[d * 4 + h];
        atomicAdd(&acc[(size_t)d * 512 + h * HID_ + c],
                  alpha * xp[(size_t)s * 512 + h * HID_ + c]);
    }
}

// -------- head-mean + bias + relu -> bf16 into xcat[:, 0:128] --------------
__global__ void k_gat_finalize(const float* __restrict__ acc,
                               const float* __restrict__ gbias,
                               unsigned short* __restrict__ xcat) {
    int n = blockIdx.x, c = threadIdx.x;   // 128 threads
    const float* a = acc + (size_t)n * 512;
    float v = (a[c] + a[HID_ + c] + a[2 * HID_ + c] + a[3 * HID_ + c]) * 0.25f + gbias[c];
    xcat[(size_t)n * 256 + c] = f2bf(fmaxf(v, 0.f));
}

// -------- per-node 4x4 cross attention -> bf16 into xcat[:, 128:256] -------
__global__ void k_cross_attn(const float* __restrict__ q, const float* __restrict__ kv,
                             unsigned short* __restrict__ xcat) {
    __shared__ float qs[128], ks[128], vs[128], sc[16], at[16];
    int n = blockIdx.x, t = threadIdx.x;   // 128 threads
    qs[t] = q[(size_t)n * 128 + t];
    ks[t] = kv[(size_t)n * 256 + t];
    vs[t] = kv[(size_t)n * 256 + 128 + t];
    __syncthreads();
    if (t < 16) {
        int h = t >> 2, g = t & 3;
        float s = 0.f;
#pragma unroll
        for (int d = 0; d < 32; ++d) s += qs[h * 32 + d] * ks[g * 32 + d];
        sc[t] = s * 0.17677669529663687f;  // 1/sqrt(32)
    }
    __syncthreads();
    if (t < 4) {
        float a0 = sc[t*4], a1 = sc[t*4+1], a2 = sc[t*4+2], a3 = sc[t*4+3];
        float m = fmaxf(fmaxf(a0, a1), fmaxf(a2, a3));
        float e0 = __expf(a0-m), e1 = __expf(a1-m), e2 = __expf(a2-m), e3 = __expf(a3-m);
        float inv = 1.f / (e0 + e1 + e2 + e3);
        at[t*4] = e0*inv; at[t*4+1] = e1*inv; at[t*4+2] = e2*inv; at[t*4+3] = e3*inv;
    }
    __syncthreads();
    int h = t >> 5, d = t & 31;
    float o = at[h*4+0]*vs[d] + at[h*4+1]*vs[32+d] + at[h*4+2]*vs[64+d] + at[h*4+3]*vs[96+d];
    xcat[(size_t)n * 256 + 128 + t] = f2bf(o);
}

// ---- final GEMM (xcat @ Wf + bf) fused with LayerNorm + ReLU -> d_out -----
// 8 waves: wave w owns N-tile w; the block owns full 16x128 rows in LDS.
__global__ void k_gemm_fused_ln(const unsigned short* __restrict__ A, // xcat, lda=256
                                const unsigned short* __restrict__ Wp,
                                const float* __restrict__ bias,
                                const float* __restrict__ ln_g,
                                const float* __restrict__ ln_b,
                                float* __restrict__ out) {
    __shared__ float tile[16][132];
    __shared__ float mu[16], rsg[16];
    const int t = threadIdx.x;
    const int w = t >> 5, lane = t & 31;
    const int hs = lane >> 4, lid = lane & 15;
    const int tileM = blockIdx.x;
    const unsigned short* Arow = A + (size_t)(tileM * 16 + lid) * 256;
    const unsigned short* Bp   = Wp + ((size_t)w * 32 + lane) * 16;
    v8f acc = {};
#pragma unroll
    for (int kb = 0; kb < 256; kb += 32) {
        v8bf lo = *(const v8bf*)(Arow + kb + hs * 8);
        v8bf hi = *(const v8bf*)(Arow + kb + hs * 8 + 16);
        v16bf a = __builtin_shufflevector(lo, hi, 0,1,2,3,4,5,6,7,8,9,10,11,12,13,14,15);
        v16bf b = *(const v16bf*)Bp;
        Bp += 8 * 512;
        acc = __builtin_amdgcn_wmma_f32_16x16x32_bf16(false, a, false, b,
                                                      (short)0, acc, false, false);
    }
    int col = w * 16 + lid;
    float bv = bias[col];
#pragma unroll
    for (int r = 0; r < 8; ++r) tile[r + hs * 8][col] = acc[r] + bv;
    __syncthreads();
    if (t < 16) {
        float s = 0.f, s2 = 0.f;
        for (int c = 0; c < 128; ++c) { float v = tile[t][c]; s += v; s2 += v * v; }
        float m = s * (1.f / 128.f);
        float var = s2 * (1.f / 128.f) - m * m;
        mu[t] = m; rsg[t] = rsqrtf(var + 1e-5f);
    }
    __syncthreads();
    int row = t >> 4, cb = t & 15;
    float m = mu[row], r_ = rsg[row];
#pragma unroll
    for (int j = 0; j < 8; ++j) {
        int c = cb + 16 * j;
        float v = (tile[row][c] - m) * r_ * ln_g[c] + ln_b[c];
        out[(size_t)(tileM * 16 + row) * 128 + c] = fmaxf(v, 0.f);
    }
}

// ---------------------------------------------------------------------------
extern "C" void kernel_launch(void* const* d_in, const int* in_sizes, int n_in,
                              void* d_out, int out_size, void* d_ws, size_t ws_size,
                              hipStream_t stream) {
    const float* x    = (const float*)d_in[0];
    const float* cb   = (const float*)d_in[1];
    const int*   ei   = (const int*)  d_in[2];
    const float* Wg   = (const float*)d_in[3];
    const float* att_src = (const float*)d_in[4];
    const float* att_dst = (const float*)d_in[5];
    const float* gat_bias= (const float*)d_in[6];
    const float* Wq = (const float*)d_in[7];  const float* bq = (const float*)d_in[8];
    const float* Wk = (const float*)d_in[9];  const float* bk = (const float*)d_in[10];
    const float* Wv = (const float*)d_in[11]; const float* bv = (const float*)d_in[12];
    const float* Wf = (const float*)d_in[13]; const float* bf_ = (const float*)d_in[14];
    const float* ln_g = (const float*)d_in[15];
    const float* ln_b = (const float*)d_in[16];
    float* out = (float*)d_out;

    const int N = in_sizes[0] / HID_;     // 20000
    const int E = in_sizes[2] / 2;        // 160000
    const int Etot = E + N;
    const int* esrc = ei;
    const int* edst = ei + E;

    // ------- workspace bump allocator (all offsets 256B aligned) -------
    char* p = (char*)d_ws;
    auto alloc = [&](size_t bytes) {
        char* r = p;
        p += (bytes + 255) & ~(size_t)255;
        return (void*)r;
    };
    float*          xp     = (float*)         alloc((size_t)N * 512 * 4);
    float*          accum  = (float*)         alloc((size_t)N * 512 * 4);
    unsigned short* cb_bf  = (unsigned short*)alloc((size_t)N * CB_ * 2);
    unsigned short* x_bf   = (unsigned short*)alloc((size_t)N * HID_ * 2);
    float*          kv     = (float*)         alloc((size_t)N * 256 * 4);
    float*          qbuf   = (float*)         alloc((size_t)N * 128 * 4);
    unsigned short* xcat   = (unsigned short*)alloc((size_t)N * 256 * 2);
    float*          as_    = (float*)         alloc((size_t)N * 4 * 4);
    float*          ad_    = (float*)         alloc((size_t)N * 4 * 4);
    float*          eval   = (float*)         alloc((size_t)Etot * 4 * 4);
    unsigned*       emax   = (unsigned*)      alloc((size_t)N * 4 * 4);
    float*          denom  = (float*)         alloc((size_t)N * 4 * 4);
    unsigned short* WgP    = (unsigned short*)alloc((size_t)HID_ * 512 * 2);
    unsigned short* WkvP   = (unsigned short*)alloc((size_t)CB_ * 256 * 2);
    unsigned short* WqP    = (unsigned short*)alloc((size_t)HID_ * HID_ * 2);
    unsigned short* WfP    = (unsigned short*)alloc((size_t)256 * HID_ * 2);
    float*          bkv    = (float*)         alloc(256 * 4);
    (void)ws_size;

    // ------- zero/neutral init (hipMemsetAsync is graph-capturable) -------
    hipMemsetAsync(accum, 0, (size_t)N * 512 * 4, stream);
    hipMemsetAsync(emax,  0, (size_t)N * 4 * 4, stream);   // 0 < ord(any finite)
    hipMemsetAsync(denom, 0, (size_t)N * 4 * 4, stream);

    // ------- stage 0: convert/pack -------
    long nx = (long)N * HID_, ncb = (long)N * CB_;
    k_cvt_bf16<<<dim3((unsigned)((nx  + 255) / 256)), 256, 0, stream>>>(x,  x_bf,  nx);
    k_cvt_bf16<<<dim3((unsigned)((ncb + 255) / 256)), 256, 0, stream>>>(cb, cb_bf, ncb);
    k_pack_w<<<dim3((HID_*512 + 255)/256), 256, 0, stream>>>(Wg, Wg, HID_, 512, 0, WgP);
    k_pack_w<<<dim3((CB_*256  + 255)/256), 256, 0, stream>>>(Wk, Wv, CB_, HID_, HID_, WkvP);
    k_pack_w<<<dim3((HID_*HID_+ 255)/256), 256, 0, stream>>>(Wq, Wq, HID_, HID_, 0, WqP);
    k_pack_w<<<dim3((256*HID_ + 255)/256), 256, 0, stream>>>(Wf, Wf, 256, HID_, 0, WfP);
    k_cat_bias<<<1, 256, 0, stream>>>(bk, bv, bkv);

    // ------- stage 1: xp = x @ Wg (WMMA, 16x64 tile per wave) -------
    k_gemm_bf16<<<dim3(N/16, 512/64), 32, 0, stream>>>(x_bf, HID_, WgP, nullptr,
                                                       xp, 512, HID_, 512);
    // ------- stage 2: GAT edge softmax + scatter -------
    k_attn_coef<<<dim3(N), 128, 0, stream>>>(xp, att_src, att_dst, as_, ad_);
    k_edge_logits<<<dim3((Etot + 255)/256), 256, 0, stream>>>(esrc, edst, E, Etot,
                                                              as_, ad_, eval, emax);
    k_edge_exp<<<dim3((Etot + 255)/256), 256, 0, stream>>>(esrc, edst, E, Etot,
                                                           emax, eval, denom);
    k_edge_scatter<<<dim3(Etot), 128, 0, stream>>>(esrc, edst, E, eval, denom, xp, accum);
    k_gat_finalize<<<dim3(N), 128, 0, stream>>>(accum, gat_bias, xcat);

    // ------- stage 3: projections (WMMA, 16x64 tile per wave) -------
    k_gemm_bf16<<<dim3(N/16, 256/64), 32, 0, stream>>>(cb_bf, CB_, WkvP, bkv,
                                                       kv, 256, CB_, 256);
    k_gemm_bf16<<<dim3(N/16, 128/64), 32, 0, stream>>>(xcat, 256, WqP, bq,
                                                       qbuf, 128, HID_, 128);
    // ------- stage 4: per-node cross attention -------
    k_cross_attn<<<dim3(N), 128, 0, stream>>>(qbuf, kv, xcat);

    // ------- stage 5: fusion GEMM + LayerNorm + ReLU (WMMA, fused epilogue) ----
    k_gemm_fused_ln<<<dim3(N/16), 256, 0, stream>>>(xcat, WfP, bf_, ln_g, ln_b, out);
}
